// DifferentiableRenderer_54357106098254
// MI455X (gfx1250) — compile-verified, matrix operand order checked by reading, not run on hardware
//
#include <hip/hip_runtime.h>

typedef __attribute__((ext_vector_type(16))) _Float16 v16h;
typedef __attribute__((ext_vector_type(8)))  float    v8f;

#define TILE_W   64
#define N_GAUSS  16384
#define GCHUNK   128               // gaussians staged per iteration (== blockDim)
#define KC       32                // K per WMMA
#define REC      12                // floats per LDS record: 8 payload + 4 pad (48B: b128-aligned, de-banks K halves)
#define NITER    (N_GAUSS / GCHUNK)
#define EPSV     1e-5f
#define LOG2E    1.44269504088896340736f

// ---------------- pack kernel: per-(image,gaussian) 32B record into d_ws ----------------
__global__ __launch_bounds__(256)
void gsplat_pack(const float* __restrict__ means, const float* __restrict__ covs,
                 const float* __restrict__ colors, const float* __restrict__ extr,
                 const int* __restrict__ image_ids, float* __restrict__ pk, int n_imgs)
{
    const int gid   = blockIdx.x * blockDim.x + threadIdx.x;
    const int total = n_imgs * N_GAUSS;
    if (gid >= total) return;
    const int img = gid / N_GAUSS;
    const int g   = gid - img * N_GAUSS;
    const int cam = image_ids[img];
    const float* E = extr + cam * 16;
    const float ox = E[3], oy = E[7], oz = E[11];
    const float oo = ox*ox + oy*oy + oz*oz;
    const float mx = means[g*3+0], my = means[g*3+1], mz = means[g*3+2];
    const float s2 = (covs[g*9+0] + covs[g*9+4] + covs[g*9+8]) * (1.0f/3.0f);
    const float base = mx*mx + my*my + mz*mz - 2.0f*(mx*ox + my*oy + mz*oz) + oo;
    const float fco  = -0.5f * LOG2E / (s2 + EPSV);
    float4* P = (float4*)(pk + (size_t)gid * 8);
    P[0] = make_float4(mx, my, mz, base);
    P[1] = make_float4(fco, colors[g*3+0], colors[g*3+1], colors[g*3+2]);
}

// ---------------- shared inner loop: 4 WMMA K-steps over one staged chunk ----------------
__device__ __forceinline__ void accum_chunks(const float* __restrict__ shb,
                                             float dX, float dY, float dZ,
                                             float nodotd, float coef,
                                             float selr, float selg, float selb, float bias,
                                             int hi, v8f& acc)
{
    for (int c = 0; c < GCHUNK; c += KC) {
        const float* shc = shb + c * REC;
        v16h A, B;
#pragma unroll
        for (int v = 0; v < 8; ++v) {
#pragma unroll
            for (int h = 0; h < 2; ++h) {
                // ISA 16-bit A layout: lanes 0-15 K{0-7,16-23}, lanes 16-31 K{8-15,24-31}
                const int k = (v < 4) ? (8*hi + 2*v + h) : (16 + 8*hi + 2*(v-4) + h);
                const float4 q0 = *(const float4*)(shc + k*REC);      // mx,my,mz,base (broadcast)
                const float4 q1 = *(const float4*)(shc + k*REC + 4);  // f, cr,cg,cb
                float t = __builtin_fmaf(q0.x, dX,
                          __builtin_fmaf(q0.y, dY,
                          __builtin_fmaf(q0.z, dZ, nodotd)));
                float dist2 = __builtin_fmaf(-coef, t*t, q0.w);
                float w = __builtin_amdgcn_exp2f(q1.x * dist2);
                A[2*v + h] = (_Float16)w;
                // branch-free B select: column ncol -> rgb channel, col 3 -> 1, else 0
                float bv = __builtin_fmaf(q1.y, selr,
                           __builtin_fmaf(q1.z, selg,
                           __builtin_fmaf(q1.w, selb, bias)));
                B[2*v + h] = (_Float16)bv;
            }
        }
        acc = __builtin_amdgcn_wmma_f32_16x16x32_f16(false, A, false, B,
                                                     (short)0, acc, false, false);
    }
}

__device__ __forceinline__ void writeback(const v8f& acc, int lane, int wave, int hi,
                                          int ncol, int blockRow, int img,
                                          float* __restrict__ out)
{
#pragma unroll
    for (int j = 0; j < 8; ++j) {
        float v    = acc[j];
        float wsum = __shfl(v, (lane & 16) | 3, 32);   // D column 3 = row weight-sum
        if (ncol < 3) {
            const int x = wave*16 + j + hi*8;
            const int p = blockRow * TILE_W + x;
            out[(img*3 + ncol) * (TILE_W*TILE_W) + p] = v / (wsum + EPSV);
        }
    }
}

// ---------------- render, async-DMA staging from packed workspace ----------------
__global__ __launch_bounds__(128)
void gsplat_render_async(const float* __restrict__ pk,
                         const float* __restrict__ intr,
                         const float* __restrict__ extr,
                         const int*   __restrict__ image_ids,
                         float* __restrict__ out)
{
    __shared__ __align__(16) float sh[2][GCHUNK * REC];

    const int tid  = threadIdx.x;
    const int lane = tid & 31;
    const int wave = tid >> 5;
    const int hi   = (lane >> 4) & 1;
    const int ncol = lane & 15;

    const int blockRow = blockIdx.x & (TILE_W - 1);
    const int img      = blockIdx.x / TILE_W;
    const int cam      = image_ids[img];

    const float fx = intr[cam*4+0], fy = intr[cam*4+1];
    const float cx = intr[cam*4+2], cy = intr[cam*4+3];
    const float* E = extr + cam*16;
    const float ox = E[3], oy = E[7], oz = E[11];

    const float xf = (float)(wave*16 + ncol);
    const float yf = (float)blockRow;
    const float dirx =  (xf - cx)/fx, diry = -(yf - cy)/fy, dirz = -1.0f;
    const float dX = E[0]*dirx + E[1]*diry + E[2]*dirz;
    const float dY = E[4]*dirx + E[5]*diry + E[6]*dirz;
    const float dZ = E[8]*dirx + E[9]*diry + E[10]*dirz;
    const float nodotd = -(ox*dX + oy*dY + oz*dZ);
    const float coef   = 2.0f - (dX*dX + dY*dY + dZ*dZ);

    const float selr = (ncol == 0) ? 1.0f : 0.0f;
    const float selg = (ncol == 1) ? 1.0f : 0.0f;
    const float selb = (ncol == 2) ? 1.0f : 0.0f;
    const float bias = (ncol == 3) ? 1.0f : 0.0f;

    const float* src0 = pk + (size_t)img * N_GAUSS * 8;

    auto stage = [&](int buf, int g0) {
        const float* src = src0 + (size_t)(g0 + tid) * 8;
        unsigned dst = (unsigned)(unsigned long long)(uintptr_t)&sh[buf][tid * REC];
        // DMA 32B record straight into LDS (ASYNCcnt); INST_OFFSET applies to both sides
        asm volatile("global_load_async_to_lds_b128 %0, %1, off\n\t"
                     "global_load_async_to_lds_b128 %0, %1, off offset:16"
                     :: "v"(dst), "v"(src) : "memory");
    };

    stage(0, 0);
    asm volatile("s_wait_asynccnt 0x0" ::: "memory");
    __syncthreads();

    v8f acc = {};
    for (int it = 0; it < NITER; ++it) {
        const int cur = it & 1;
        if (it + 1 < NITER) stage(cur ^ 1, (it + 1) * GCHUNK);   // DMA next chunk, overlapped
        accum_chunks(sh[cur], dX, dY, dZ, nodotd, coef, selr, selg, selb, bias, hi, acc);
        asm volatile("s_wait_asynccnt 0x0" ::: "memory");
        __syncthreads();
    }

    writeback(acc, lane, wave, hi, ncol, blockRow, img, out);
}

// ---------------- render, direct staging fallback (workspace too small) ----------------
__global__ __launch_bounds__(128)
void gsplat_render_direct(const float* __restrict__ means,
                          const float* __restrict__ covs,
                          const float* __restrict__ colors,
                          const float* __restrict__ intr,
                          const float* __restrict__ extr,
                          const int*   __restrict__ image_ids,
                          float* __restrict__ out)
{
    __shared__ __align__(16) float sh[2][GCHUNK * REC];

    const int tid  = threadIdx.x;
    const int lane = tid & 31;
    const int wave = tid >> 5;
    const int hi   = (lane >> 4) & 1;
    const int ncol = lane & 15;

    const int blockRow = blockIdx.x & (TILE_W - 1);
    const int img      = blockIdx.x / TILE_W;
    const int cam      = image_ids[img];

    const float fx = intr[cam*4+0], fy = intr[cam*4+1];
    const float cx = intr[cam*4+2], cy = intr[cam*4+3];
    const float* E = extr + cam*16;
    const float ox = E[3], oy = E[7], oz = E[11];
    const float oo = ox*ox + oy*oy + oz*oz;

    const float xf = (float)(wave*16 + ncol);
    const float yf = (float)blockRow;
    const float dirx =  (xf - cx)/fx, diry = -(yf - cy)/fy, dirz = -1.0f;
    const float dX = E[0]*dirx + E[1]*diry + E[2]*dirz;
    const float dY = E[4]*dirx + E[5]*diry + E[6]*dirz;
    const float dZ = E[8]*dirx + E[9]*diry + E[10]*dirz;
    const float nodotd = -(ox*dX + oy*dY + oz*dZ);
    const float coef   = 2.0f - (dX*dX + dY*dY + dZ*dZ);

    const float selr = (ncol == 0) ? 1.0f : 0.0f;
    const float selg = (ncol == 1) ? 1.0f : 0.0f;
    const float selb = (ncol == 2) ? 1.0f : 0.0f;
    const float bias = (ncol == 3) ? 1.0f : 0.0f;

    // prologue: stage chunk 0
    {
        const int g = tid;
        const float mx = means[g*3+0], my = means[g*3+1], mz = means[g*3+2];
        const float s2 = (covs[g*9+0] + covs[g*9+4] + covs[g*9+8]) * (1.0f/3.0f);
        const float base = mx*mx + my*my + mz*mz - 2.0f*(mx*ox + my*oy + mz*oz) + oo;
        const float fco  = -0.5f * LOG2E / (s2 + EPSV);
        *(float4*)&sh[0][tid*REC]     = make_float4(mx, my, mz, base);
        *(float4*)&sh[0][tid*REC + 4] = make_float4(fco, colors[g*3+0], colors[g*3+1], colors[g*3+2]);
    }
    __syncthreads();

    v8f acc = {};
    for (int it = 0; it < NITER; ++it) {
        const int cur = it & 1;
        const bool has_next = (it + 1 < NITER);
        float mx = 0, my = 0, mz = 0, c0 = 0, c4 = 0, c8 = 0, r = 0, g1 = 0, b1 = 0;
        if (has_next) {                            // issue next chunk's global loads first
            const int g = (it + 1) * GCHUNK + tid;
            mx = means[g*3+0]; my = means[g*3+1]; mz = means[g*3+2];
            c0 = covs[g*9+0];  c4 = covs[g*9+4];  c8 = covs[g*9+8];
            r  = colors[g*3+0]; g1 = colors[g*3+1]; b1 = colors[g*3+2];
            __builtin_prefetch(&means[(g + GCHUNK) * 3], 0, 0);
        }
        accum_chunks(sh[cur], dX, dY, dZ, nodotd, coef, selr, selg, selb, bias, hi, acc);
        if (has_next) {
            const float s2   = (c0 + c4 + c8) * (1.0f/3.0f);
            const float base = mx*mx + my*my + mz*mz - 2.0f*(mx*ox + my*oy + mz*oz) + oo;
            const float fco  = -0.5f * LOG2E / (s2 + EPSV);
            *(float4*)&sh[cur ^ 1][tid*REC]     = make_float4(mx, my, mz, base);
            *(float4*)&sh[cur ^ 1][tid*REC + 4] = make_float4(fco, r, g1, b1);
        }
        __syncthreads();
    }

    writeback(acc, lane, wave, hi, ncol, blockRow, img, out);
}

extern "C" void kernel_launch(void* const* d_in, const int* in_sizes, int n_in,
                              void* d_out, int out_size, void* d_ws, size_t ws_size,
                              hipStream_t stream) {
    const float* means   = (const float*)d_in[0];
    const float* covs    = (const float*)d_in[1];
    const float* colors  = (const float*)d_in[2];
    const float* intr    = (const float*)d_in[3];
    const float* extr    = (const float*)d_in[4];
    const int*   ids     = (const int*)  d_in[5];
    float*       out     = (float*)d_out;

    const int    n_imgs = in_sizes[5];
    const size_t need   = (size_t)n_imgs * N_GAUSS * 8 * sizeof(float);
    dim3 rgrid(n_imgs * TILE_W), rblock(128);

    if (d_ws && ws_size >= need) {
        const int total = n_imgs * N_GAUSS;
        hipLaunchKernelGGL(gsplat_pack, dim3((total + 255) / 256), dim3(256), 0, stream,
                           means, covs, colors, extr, ids, (float*)d_ws, n_imgs);
        hipLaunchKernelGGL(gsplat_render_async, rgrid, rblock, 0, stream,
                           (const float*)d_ws, intr, extr, ids, out);
    } else {
        hipLaunchKernelGGL(gsplat_render_direct, rgrid, rblock, 0, stream,
                           means, covs, colors, intr, extr, ids, out);
    }
}
